// SAGEEncode_54863912239192
// MI455X (gfx1250) — matrix operand order repaired and checked by reading, not compiled, not run on hardware
//
#include <hip/hip_runtime.h>
#include <hip/hip_bf16.h>

// ---------------------------------------------------------------------------
// SAGE encoder for MI455X (gfx1250): bf16 WMMA GEMMs, fp32 accumulate.
//   B=2048, F1=10, F2=25, D=128, H=256, N=261
//   GEMM1: (22528 x 256) @ W1(256x256) + b1, ReLU
//   GEMM2: ( 2048 x 512) @ W2(512x256) + b2
// Memory-bound: ~350 MB total traffic ~= 15 us at 23.3 TB/s; GEMM compute
// (3.5 GFLOP) is negligible on the bf16 WMMA path.
// ---------------------------------------------------------------------------

#define SAGE_B   2048
#define SAGE_F1  10
#define SAGE_F2  25
#define SAGE_D   128
#define SAGE_H   256
#define SAGE_N   (1 + SAGE_F1 + SAGE_F1 * SAGE_F2)   // 261
#define R1       (SAGE_B * (SAGE_F1 + 1))            // 22528 rows for GEMM1
#define K1       (2 * SAGE_D)                        // 256
#define K2       (2 * SAGE_H)                        // 512

typedef __attribute__((ext_vector_type(16))) __bf16      v16bf;
typedef __attribute__((ext_vector_type(8)))  float       v8f;
typedef __attribute__((ext_vector_type(4)))  unsigned int v4u;

union bfrag {
    v4u   u[2];   // two 16-byte chunks = 16 bf16 values
    v16bf h;
};

__device__ __forceinline__ unsigned short f2bf(float f) {
    unsigned int u = __float_as_uint(f);
    // round-to-nearest-even truncation to bf16
    u += 0x7FFFu + ((u >> 16) & 1u);
    return (unsigned short)(u >> 16);
}

// ---------------------------------------------------------------------------
// Kernel 1: neighbor means + pack GEMM1 A-matrix (bf16, row-major [R1][256]).
// One block per sample b, 128 threads (one per feature d). All x reads
// coalesced. Row r = b*11 + j:
//   j==0 -> [seed | mean(hop1)],  j>=1 -> [hop1_j | mean(hop2_j)]
// ---------------------------------------------------------------------------
__global__ __launch_bounds__(128) void pack_a1(const float* __restrict__ x,
                                               unsigned short* __restrict__ A1) {
    const int b = blockIdx.x;
    const int d = threadIdx.x;                    // 0..127
    const float* xb = x + (size_t)b * SAGE_N * SAGE_D;
    const size_t r0 = (size_t)b * (SAGE_F1 + 1) * K1;

    // seed row
    float seed = xb[d];
    float s1 = 0.f;
#pragma unroll
    for (int k = 0; k < SAGE_F1; ++k) s1 += xb[(1 + k) * SAGE_D + d];
    A1[r0 + d]          = f2bf(seed);
    A1[r0 + SAGE_D + d] = f2bf(s1 * (1.0f / SAGE_F1));

    // hop1 rows
    for (int j = 1; j <= SAGE_F1; ++j) {
        float src = xb[j * SAGE_D + d];
        const float* h2 = xb + (size_t)(1 + SAGE_F1 + (j - 1) * SAGE_F2) * SAGE_D;
        float s2 = 0.f;
#pragma unroll
        for (int m = 0; m < SAGE_F2; ++m) s2 += h2[m * SAGE_D + d];
        A1[r0 + (size_t)j * K1 + d]          = f2bf(src);
        A1[r0 + (size_t)j * K1 + SAGE_D + d] = f2bf(s2 * (1.0f / SAGE_F2));
    }
}

// ---------------------------------------------------------------------------
// Kernel 2: repack weights W[K][N] (fp32 row-major) into per-lane WMMA B
// fragment order (bf16), kc-major so the GEMM K-loop walks a single base
// pointer with compile-time immediate offsets:
//   P = ((kc*16 + nt)*32 + lane)*16 + e
//   value = W[kc*32 + 8*(lane>>4) + (e&7) + 16*(e>>3)][nt*16 + (lane&15)]
// For a fixed wave strip (nt = nstrip*4 + t) and lane, fragment (kc, t) sits
// at byte offset kc*16384 + t*1024 from the wave's base.
// ---------------------------------------------------------------------------
__global__ __launch_bounds__(256) void pack_w(const float* __restrict__ W,
                                              unsigned short* __restrict__ Wp,
                                              int K, int N) {
    int tid = blockIdx.x * blockDim.x + threadIdx.x;
    if (tid >= K * N) return;
    int e    = tid & 15;
    int rest = tid >> 4;
    int lane = rest & 31;
    rest >>= 5;
    int nt = rest & 15;
    int kc = rest >> 4;
    int k = kc * 32 + 8 * (lane >> 4) + (e & 7) + 16 * (e >> 3);
    int n = nt * 16 + (lane & 15);
    Wp[tid] = f2bf(W[(size_t)k * N + n]);
}

// ---------------------------------------------------------------------------
// Kernel 3: bf16 WMMA GEMM.  C[R][256] = A[R][K] @ W[K][256] + bias (opt ReLU)
// One wave computes a 16x64 output strip (4 accumulator tiles).
// blockDim = 128 (4 waves); wave w covers columns [w*64, w*64+64).
// gridDim.x = R/16.  K and RELU are compile-time: the K-loop fully unrolls
// and every A/B fragment load becomes global_load_b128 with an immediate
// offset from one of two base pointers (no per-iteration address VALU).
// A fragment (16-bit 16x32, ISA 7.12.2): lane l, m=l&15, half=l>>4:
//   VGPR0..3 = A[m][k0 + 8*half + 0..7], VGPR4..7 = A[m][k0 + 16 + 8*half + 0..7]
// ---------------------------------------------------------------------------
template <int K, int RELU>
__global__ __launch_bounds__(128) void gemm_bf16(const unsigned short* __restrict__ A,
                                                 const unsigned short* __restrict__ Wp,
                                                 const float* __restrict__ bias,
                                                 float* __restrict__ C) {
    constexpr int KC = K / 32;
    const int lane = threadIdx.x & 31;
    const int wave = threadIdx.x >> 5;      // column strip 0..3
    const int m0   = blockIdx.x << 4;       // row-tile base
    const int half = lane >> 4;
    const int ml   = lane & 15;             // A row within tile / C column within tile

    v8f acc[4];
#pragma unroll
    for (int t = 0; t < 4; ++t) acc[t] = (v8f)(0.0f);

    const unsigned short* ap = A + (size_t)(m0 + ml) * K + 8 * half;
    // base for (kc=0, t=0) of this wave's strip
    const unsigned short* bp = Wp + ((size_t)((wave * 4) * 32 + lane) << 4);

#pragma unroll
    for (int kc = 0; kc < KC; ++kc) {
        bfrag a;
        a.u[0] = *(const v4u*)(ap + kc * 32);
        a.u[1] = *(const v4u*)(ap + kc * 32 + 16);
#pragma unroll
        for (int t = 0; t < 4; ++t) {
            const unsigned short* p = bp + kc * 8192 + t * 512;  // ushort offsets
            bfrag b;
            b.u[0] = *(const v4u*)(p);
            b.u[1] = *(const v4u*)(p + 8);
            acc[t] = __builtin_amdgcn_wmma_f32_16x16x32_bf16(
                false, a.h, false, b.h, (short)0, acc[t], false, false);
        }
    }

    // Epilogue. C/D layout: lane l, vgpr v -> row m0 + v + 8*half, col nt*16 + (l&15)
#pragma unroll
    for (int t = 0; t < 4; ++t) {
        const int n  = (wave * 4 + t) * 16 + ml;
        const float bv = bias[n];
#pragma unroll
        for (int v = 0; v < 8; ++v) {
            const int r = m0 + v + 8 * half;
            float val = acc[t][v] + bv;
            if (RELU) val = fmaxf(val, 0.0f);
            C[(size_t)r * SAGE_H + n] = val;
        }
    }
}

// ---------------------------------------------------------------------------
// Kernel 4: build GEMM2 A-matrix (bf16 [2048][512]): [h0_b | mean_j h1_b[j]]
// H layout: row b*11 + j (j=0 is h0, j=1..10 are h1 rows), 256 fp32 each.
// ---------------------------------------------------------------------------
__global__ __launch_bounds__(256) void build_a2(const float* __restrict__ H,
                                                unsigned short* __restrict__ A2) {
    const int b = blockIdx.x;
    const int n = threadIdx.x;               // 0..255
    const float* hb = H + (size_t)b * (SAGE_F1 + 1) * SAGE_H;
    A2[(size_t)b * K2 + n] = f2bf(hb[n]);
    float s = 0.f;
#pragma unroll
    for (int j = 1; j <= SAGE_F1; ++j) s += hb[(size_t)j * SAGE_H + n];
    A2[(size_t)b * K2 + SAGE_H + n] = f2bf(s * (1.0f / SAGE_F1));
}

// ---------------------------------------------------------------------------
// Launch
// ---------------------------------------------------------------------------
extern "C" void kernel_launch(void* const* d_in, const int* in_sizes, int n_in,
                              void* d_out, int out_size, void* d_ws, size_t ws_size,
                              hipStream_t stream) {
    const float* x  = (const float*)d_in[0];   // (B, 261, 128)
    const float* W1 = (const float*)d_in[1];   // (256, 256)
    const float* b1 = (const float*)d_in[2];   // (256)
    const float* W2 = (const float*)d_in[3];   // (512, 256)
    const float* b2 = (const float*)d_in[4];   // (256)
    float* out = (float*)d_out;                // (B, 1, 256)

    char* ws = (char*)d_ws;
    size_t off = 0;
    unsigned short* A1  = (unsigned short*)(ws + off); off += (size_t)R1 * K1 * 2;        // 11.5 MB
    unsigned short* W1p = (unsigned short*)(ws + off); off += (size_t)K1 * SAGE_H * 2;    // 128 KB
    unsigned short* W2p = (unsigned short*)(ws + off); off += (size_t)K2 * SAGE_H * 2;    // 256 KB
    float*          Hbuf= (float*)(ws + off);          off += (size_t)R1 * SAGE_H * 4;    // 23 MB
    unsigned short* A2  = (unsigned short*)(ws + off); off += (size_t)SAGE_B * K2 * 2;    // 2 MB
    (void)off; (void)ws_size; (void)in_sizes; (void)n_in; (void)out_size;

    // 1) neighbor means + A1 pack (reads all of x once, coalesced)
    pack_a1<<<SAGE_B, 128, 0, stream>>>(x, A1);

    // 2) weight repack into kc-major WMMA B-fragment order
    pack_w<<<(K1 * SAGE_H + 255) / 256, 256, 0, stream>>>(W1, W1p, K1, SAGE_H);
    pack_w<<<(K2 * SAGE_H + 255) / 256, 256, 0, stream>>>(W2, W2p, K2, SAGE_H);

    // 3) GEMM1: (22528 x 256) @ (256 x 256) + b1, ReLU -> Hbuf
    gemm_bf16<K1, 1><<<R1 / 16, 128, 0, stream>>>(A1, W1p, b1, Hbuf);

    // 4) h0 | mean(h1) -> A2
    build_a2<<<SAGE_B, 256, 0, stream>>>(Hbuf, A2);

    // 5) GEMM2: (2048 x 512) @ (512 x 256) + b2 -> out
    gemm_bf16<K2, 0><<<SAGE_B / 16, 128, 0, stream>>>(A2, W2p, b2, out);
}
